// MultiHeadSelfAttention_65386582114485
// MI455X (gfx1250) — matrix operand order repaired and checked by reading, not compiled, not run on hardware
//
#include <hip/hip_runtime.h>

#define D_MODEL 1024
#define NUM_HEADS 16
#define SEQ 2048
#define BATCH 2
#define DK 64
#define NTOK (BATCH * SEQ) /* 4096 */

typedef __bf16 bf16;
typedef bf16 v16bf __attribute__((ext_vector_type(16)));
typedef bf16 v8bf  __attribute__((ext_vector_type(8)));
typedef float v8f  __attribute__((ext_vector_type(8)));

static __device__ inline v16bf join8(v8bf a, v8bf b) {
  return __builtin_shufflevector(a, b, 0,1,2,3,4,5,6,7,8,9,10,11,12,13,14,15);
}

// A fragment: 16(M) x 32(K) tile from row-major source (ld in elements).
// Lanes 0-15 row M with K halves base 0, lanes 16-31 same rows with K halves base 8;
// halves 0-7 = K(base..base+7), halves 8-15 = K(base+16..base+23).
static __device__ inline v16bf load_A(const bf16* __restrict__ base, int ld, int row0, int k0) {
  const int lane = threadIdx.x & 31;
  const int r  = row0 + (lane & 15);
  const int kb = k0 + ((lane < 16) ? 0 : 8);
  const bf16* p = base + (size_t)r * ld + kb;
  v8bf a = *reinterpret_cast<const v8bf*>(p);
  v8bf b = *reinterpret_cast<const v8bf*>(p + 16);
  return join8(a, b);
}

// B fragment: 32(K) x 16(N), where B[k][n] = SRC[n][k] (SRC row-major, row = B column).
// Lane holds column n = col0 + lane%16; K base kb = k0 + (lane<16?0:8): halves 0-7 = K(kb..kb+7), halves 8-15 = K(kb+16..kb+23).
static __device__ inline v16bf load_Bt(const bf16* __restrict__ base, int ld, int col0, int k0) {
  const int lane = threadIdx.x & 31;
  const int c  = col0 + (lane & 15);
  const int kb = k0 + ((lane < 16) ? 0 : 8);
  const bf16* p = base + (size_t)c * ld + kb;
  v8bf a = *reinterpret_cast<const v8bf*>(p);
  v8bf b = *reinterpret_cast<const v8bf*>(p + 16);
  return join8(a, b);
}

static __device__ inline v8f wmma_bf16(v16bf a, v16bf b, v8f c) {
  return __builtin_amdgcn_wmma_f32_16x16x32_bf16(false, a, false, b, (short)0, c, false, false);
}

static __device__ inline v8f vzero8() {
  v8f z = {0.f,0.f,0.f,0.f,0.f,0.f,0.f,0.f};
  return z;
}

// ---------------------------------------------------------------- stage 1: fp32 -> bf16
__global__ void cvt_f32_bf16(const float* __restrict__ in, bf16* __restrict__ out, int n) {
  for (int i = blockIdx.x * blockDim.x + threadIdx.x; i < n; i += gridDim.x * blockDim.x)
    out[i] = (bf16)in[i];
}

// ---------------------------------------------------------------- stage 2: QKV projection
// y[t,e] = sum_d X[t,d] * W[e,d]; 32 rows x 64 cols per wave (B frags reused by 2 A tiles).
__global__ __launch_bounds__(128)
void qkv_proj(const bf16* __restrict__ X,
              const bf16* __restrict__ Wq, const bf16* __restrict__ Wk, const bf16* __restrict__ Wv,
              bf16* __restrict__ Q, bf16* __restrict__ K, bf16* __restrict__ V) {
  const int wave = threadIdx.x >> 5;
  const int lm   = threadIdx.x & 15;
  const int hi8  = ((threadIdx.x & 31) < 16) ? 0 : 8;
  const int tm = blockIdx.x * 128 + wave * 32;  // 32 token rows for this wave
  const int e0 = blockIdx.y * 64;               // output-feature base
  const bf16* W  = (blockIdx.z == 0) ? Wq : (blockIdx.z == 1 ? Wk : Wv);
  bf16*       Ot = (blockIdx.z == 0) ? Q  : (blockIdx.z == 1 ? K  : V);

  v8f acc[2][4];
#pragma unroll
  for (int m = 0; m < 2; ++m)
#pragma unroll
    for (int n = 0; n < 4; ++n) acc[m][n] = vzero8();

  for (int k0 = 0; k0 < D_MODEL; k0 += 32) {
    v16bf a0 = load_A(X, D_MODEL, tm,      k0);
    v16bf a1 = load_A(X, D_MODEL, tm + 16, k0);
#pragma unroll
    for (int n = 0; n < 4; ++n) {
      v16bf b = load_Bt(W, D_MODEL, e0 + n * 16, k0);
      acc[0][n] = wmma_bf16(a0, b, acc[0][n]);
      acc[1][n] = wmma_bf16(a1, b, acc[1][n]);
    }
  }
#pragma unroll
  for (int m = 0; m < 2; ++m) {
#pragma unroll
    for (int n = 0; n < 4; ++n) {
      const int e = e0 + n * 16 + lm;
      const int h = e >> 6, dp = e & 63;
#pragma unroll
      for (int r = 0; r < 8; ++r) {
        const int t = tm + m * 16 + r + hi8;
        const int bb = t >> 11, s = t & (SEQ - 1);
        Ot[(((size_t)bb * NUM_HEADS + h) * SEQ + s) * DK + dp] = (bf16)acc[m][n][r];
      }
    }
  }
}

// ---------------------------------------------------------------- stage 3: RoPE (in-place)
__global__ void rope_kernel(bf16* __restrict__ A, const int* __restrict__ pos, int total) {
  int i = blockIdx.x * blockDim.x + threadIdx.x;
  if (i >= total) return;                 // total = B*H*S*32 pairs
  const int pair = i & 31;
  const int bhs  = i >> 5;
  const int s    = bhs & (SEQ - 1);
  const size_t base = (size_t)bhs * DK;
  const float inv = __expf((-2.0f * (float)pair / 64.0f) * 9.210340371976184f); // theta^(-2i/dk)
  const float ang = (float)pos[s] * inv;
  float sn, cs;
  __sincosf(ang, &sn, &cs);
  const float x0 = (float)A[base + 2 * pair];
  const float x1 = (float)A[base + 2 * pair + 1];
  A[base + 2 * pair]     = (bf16)(x0 * cs - x1 * sn);
  A[base + 2 * pair + 1] = (bf16)(x0 * sn + x1 * cs);
}

// ---------------------------------------------------------------- stage 4: causal flash attention
// Block = 4 waves; 64 query rows per block (16 per wave); 32-key steps.
// K block staged with GLOBAL_LOAD_ASYNC_TO_LDS_B128 (ASYNCcnt path).
__global__ __launch_bounds__(128)
void flash_attn(const bf16* __restrict__ Q, const bf16* __restrict__ K,
                const bf16* __restrict__ V, bf16* __restrict__ ctx) {
  __shared__ alignas(16) bf16 Ksh[32][72];      // [key][dk]   (+8 pad)
  __shared__ alignas(16) bf16 Vt[64][40];       // [dk][key]   (+8 pad)
  __shared__ alignas(16) bf16 Psh[4][16][40];   // per-wave P scratch (16x32, +8 pad)

  const int wave = threadIdx.x >> 5;
  const int lane = threadIdx.x & 31;
  const int lm   = lane & 15;
  const int hi8  = (lane < 16) ? 0 : 8;
  const int bh = blockIdx.y;                    // b*H + h
  const int m0 = blockIdx.x * 64;
  const int mrow = m0 + wave * 16;

  const bf16* Qh = Q + (size_t)bh * SEQ * DK;
  const bf16* Kh = K + (size_t)bh * SEQ * DK;
  const bf16* Vh = V + (size_t)bh * SEQ * DK;

  const v16bf qa0 = load_A(Qh, DK, mrow, 0);
  const v16bf qa1 = load_A(Qh, DK, mrow, 32);

  v8f o[4];
#pragma unroll
  for (int d = 0; d < 4; ++d) o[d] = vzero8();
  float mstat[8], lstat[8];
#pragma unroll
  for (int r = 0; r < 8; ++r) { mstat[r] = -INFINITY; lstat[r] = 0.0f; }

  const float scale = 0.125f;  // 1/sqrt(64)

  for (int n0 = 0; n0 < m0 + 64; n0 += 32) {
    // --- K block: async global->LDS copy. 32 rows x 64 halves = 256 x 16B chunks,
    //     2 chunks per thread (per-lane LDS dest addr handles the +8 row padding).
#pragma unroll
    for (int i = 0; i < 2; ++i) {
      const int c   = threadIdx.x + i * 128;    // chunk id 0..255
      const int row = c >> 3;                   // 8 chunks per 64-half row
      const int ch  = c & 7;
      const unsigned long long ga =
          (unsigned long long)(Kh + (size_t)(n0 + row) * DK + ch * 8);
      const unsigned la = (unsigned)(uintptr_t)&Ksh[row][ch * 8];
      asm volatile("global_load_async_to_lds_b128 %0, %1, off"
                   :: "v"(la), "v"(ga) : "memory");
    }
    // --- V block: load + transpose through VGPRs into Vt
    for (int e = threadIdx.x; e < 32 * 64; e += 128) {
      const int rr = e >> 6, d = e & 63;
      Vt[d][rr] = Vh[(size_t)(n0 + rr) * DK + d];
    }
    asm volatile("s_wait_asynccnt 0" ::: "memory");
    __syncthreads();

    if (n0 <= mrow + 15) {                      // wave-uniform: EXEC all-ones inside
      v8f s0 = vzero8(), s1 = vzero8();
      {
        v16bf b;
        b = load_Bt(&Ksh[0][0], 72,  0,  0); s0 = wmma_bf16(qa0, b, s0);
        b = load_Bt(&Ksh[0][0], 72,  0, 32); s0 = wmma_bf16(qa1, b, s0);
        b = load_Bt(&Ksh[0][0], 72, 16,  0); s1 = wmma_bf16(qa0, b, s1);
        b = load_Bt(&Ksh[0][0], 72, 16, 32); s1 = wmma_bf16(qa1, b, s1);
      }
      const bool need_mask = (n0 + 31 > mrow);
      float pr0[8], pr1[8];
#pragma unroll
      for (int r = 0; r < 8; ++r) {
        float a0 = s0[r] * scale, a1 = s1[r] * scale;
        const int row = mrow + r + hi8;
        if (need_mask) {
          if (n0 + lm > row)      a0 = -INFINITY;
          if (n0 + 16 + lm > row) a1 = -INFINITY;
        }
        float v = fmaxf(a0, a1);
        v = fmaxf(v, __shfl_xor(v, 1, 32));
        v = fmaxf(v, __shfl_xor(v, 2, 32));
        v = fmaxf(v, __shfl_xor(v, 4, 32));
        v = fmaxf(v, __shfl_xor(v, 8, 32));
        const float mnew = fmaxf(mstat[r], v);
        const float p0 = __expf(a0 - mnew);
        const float p1 = __expf(a1 - mnew);
        float ps = p0 + p1;
        ps += __shfl_xor(ps, 1, 32);
        ps += __shfl_xor(ps, 2, 32);
        ps += __shfl_xor(ps, 4, 32);
        ps += __shfl_xor(ps, 8, 32);
        const float alpha = __expf(mstat[r] - mnew);
        mstat[r] = mnew;
        lstat[r] = lstat[r] * alpha + ps;
#pragma unroll
        for (int d = 0; d < 4; ++d) o[d][r] *= alpha;
        pr0[r] = p0; pr1[r] = p1;
      }
      // P (f32 C-fragment) -> bf16 A-fragment via per-wave LDS scratch
#pragma unroll
      for (int r = 0; r < 8; ++r) {
        Psh[wave][r + hi8][lm]      = (bf16)pr0[r];
        Psh[wave][r + hi8][16 + lm] = (bf16)pr1[r];
      }
      const v16bf pa = load_A(&Psh[wave][0][0], 40, 0, 0);
#pragma unroll
      for (int d = 0; d < 4; ++d) {
        v16bf bv = load_Bt(&Vt[0][0], 40, d * 16, 0);
        o[d] = wmma_bf16(pa, bv, o[d]);
      }
    }
    __syncthreads();
  }

  // epilogue: normalize and write ctx in token-major [NTOK, D_MODEL] bf16
  const int h  = bh & (NUM_HEADS - 1);
  const int bb = bh >> 4;
#pragma unroll
  for (int d = 0; d < 4; ++d) {
#pragma unroll
    for (int r = 0; r < 8; ++r) {
      const int row = mrow + r + hi8;
      const float inv = (lstat[r] > 0.0f) ? (1.0f / lstat[r]) : 0.0f;
      const size_t t = (size_t)bb * SEQ + row;
      ctx[t * D_MODEL + h * DK + d * 16 + lm] = (bf16)(o[d][r] * inv);
    }
  }
}

// ---------------------------------------------------------------- stage 5: output projection (fp32 out)
__global__ __launch_bounds__(128)
void out_proj(const bf16* __restrict__ Ctx, const bf16* __restrict__ Wo,
              float* __restrict__ Out) {
  const int wave = threadIdx.x >> 5;
  const int lm   = threadIdx.x & 15;
  const int hi8  = ((threadIdx.x & 31) < 16) ? 0 : 8;
  const int tm = blockIdx.x * 128 + wave * 32;
  const int e0 = blockIdx.y * 64;

  v8f acc[2][4];
#pragma unroll
  for (int m = 0; m < 2; ++m)
#pragma unroll
    for (int n = 0; n < 4; ++n) acc[m][n] = vzero8();

  for (int k0 = 0; k0 < D_MODEL; k0 += 32) {
    v16bf a0 = load_A(Ctx, D_MODEL, tm,      k0);
    v16bf a1 = load_A(Ctx, D_MODEL, tm + 16, k0);
#pragma unroll
    for (int n = 0; n < 4; ++n) {
      v16bf b = load_Bt(Wo, D_MODEL, e0 + n * 16, k0);
      acc[0][n] = wmma_bf16(a0, b, acc[0][n]);
      acc[1][n] = wmma_bf16(a1, b, acc[1][n]);
    }
  }
#pragma unroll
  for (int m = 0; m < 2; ++m) {
#pragma unroll
    for (int n = 0; n < 4; ++n) {
#pragma unroll
      for (int r = 0; r < 8; ++r) {
        const size_t t = (size_t)(tm + m * 16 + r + hi8);
        Out[t * D_MODEL + e0 + n * 16 + lm] = acc[m][n][r];
      }
    }
  }
}

// ---------------------------------------------------------------- launch
extern "C" void kernel_launch(void* const* d_in, const int* in_sizes, int n_in,
                              void* d_out, int out_size, void* d_ws, size_t ws_size,
                              hipStream_t stream) {
  const float* x  = (const float*)d_in[0];
  const float* Wq = (const float*)d_in[1];
  const float* Wk = (const float*)d_in[2];
  const float* Wv = (const float*)d_in[3];
  const float* Wo = (const float*)d_in[4];
  const int*  pos = (const int*)d_in[5];
  float* out = (float*)d_out;

  char* ws = (char*)d_ws;
  const size_t XN = (size_t)NTOK * D_MODEL;        // 4M elements
  const size_t WN = (size_t)D_MODEL * D_MODEL;     // 1M elements
  size_t off = 0;
  bf16* Xb  = (bf16*)(ws + off); off += XN * 2;
  bf16* Wqb = (bf16*)(ws + off); off += WN * 2;
  bf16* Wkb = (bf16*)(ws + off); off += WN * 2;
  bf16* Wvb = (bf16*)(ws + off); off += WN * 2;
  bf16* Wob = (bf16*)(ws + off); off += WN * 2;
  bf16* Qh  = (bf16*)(ws + off); off += XN * 2;    // [B,H,S,DK]
  bf16* Kh  = (bf16*)(ws + off); off += XN * 2;
  bf16* Vh  = (bf16*)(ws + off); off += XN * 2;
  bf16* Ctx = (bf16*)(ws + off); off += XN * 2;    // [NTOK, D_MODEL]

  // stage 1: conversions
  cvt_f32_bf16<<<4096, 256, 0, stream>>>(x,  Xb,  (int)XN);
  cvt_f32_bf16<<<2048, 256, 0, stream>>>(Wq, Wqb, (int)WN);
  cvt_f32_bf16<<<2048, 256, 0, stream>>>(Wk, Wkb, (int)WN);
  cvt_f32_bf16<<<2048, 256, 0, stream>>>(Wv, Wvb, (int)WN);
  cvt_f32_bf16<<<2048, 256, 0, stream>>>(Wo, Wob, (int)WN);

  // stage 2: QKV projections (z selects Q/K/V)
  qkv_proj<<<dim3(NTOK / 128, D_MODEL / 64, 3), 128, 0, stream>>>(Xb, Wqb, Wkb, Wvb, Qh, Kh, Vh);

  // stage 3: RoPE on Q and K
  const int pairs = BATCH * NUM_HEADS * SEQ * (DK / 2);  // 2,097,152
  rope_kernel<<<(pairs + 255) / 256, 256, 0, stream>>>(Qh, pos, pairs);
  rope_kernel<<<(pairs + 255) / 256, 256, 0, stream>>>(Kh, pos, pairs);

  // stage 4: causal flash attention
  flash_attn<<<dim3(SEQ / 64, BATCH * NUM_HEADS), 128, 0, stream>>>(Qh, Kh, Vh, Ctx);

  // stage 5: output projection -> fp32
  out_proj<<<dim3(NTOK / 64, D_MODEL / 64), 128, 0, stream>>>(Ctx, Wob, out);
}
